// RNNModule_23072564314518
// MI455X (gfx1250) — matrix operand order repaired and checked by reading, not compiled, not run on hardware
//
#include <hip/hip_runtime.h>
#include <hip/hip_bf16.h>

// Problem sizes (match reference)
#define B_   64
#define T_   2048
#define DIN  128
#define H_   256
#define DOUT 128

// LDS row strides: pad by 8 bf16 (16B) so row stride == 4 DWORDs mod 64 banks
// -> 16 lanes doing b128 row loads hit 16*4 distinct banks (conflict-free).
#define LDX (DIN + 8)   // 136
#define LDH (H_ + 8)    // 264

typedef __bf16 bf16;
typedef __attribute__((ext_vector_type(16))) __bf16 v16bf;
typedef __attribute__((ext_vector_type(8)))  float  v8f;

union FragU { v16bf v; uint4 q[2]; };

// ---------------------------------------------------------------------------
// CDNA5 async global<->LDS helpers (ASYNCcnt path).
// Builtin signature (probe-derived from diagnostics): 4 args, arg0 is a
// generic `int __vector(4)*`.  LDS side passed as addrspace(3) v4i* built
// from the low 32 bits of the generic address (== LDS byte offset); AS3
// converts implicitly to generic if the builtin wants generic.
// ---------------------------------------------------------------------------
typedef int v4i __attribute__((vector_size(16)));
typedef __attribute__((address_space(3))) v4i v4i_lds;

__device__ __forceinline__ v4i_lds* to_lds_v4i(void* l) {
  return (v4i_lds*)(unsigned)(size_t)l;
}

__device__ __forceinline__ void async_g2l_b128(const void* g, void* l) {
#if __has_builtin(__builtin_amdgcn_global_load_async_to_lds_b128)
  __builtin_amdgcn_global_load_async_to_lds_b128((v4i*)g, to_lds_v4i(l), 0, 0);
#else
  unsigned lo = (unsigned)(size_t)l;
  unsigned long long ga = (unsigned long long)(size_t)g;
  asm volatile("global_load_async_to_lds_b128 %0, %1, off"
               :: "v"(lo), "v"(ga) : "memory");
#endif
}

__device__ __forceinline__ void async_l2g_b128(void* l, void* g) {
#if __has_builtin(__builtin_amdgcn_global_store_async_from_lds_b128)
  __builtin_amdgcn_global_store_async_from_lds_b128((v4i*)g, to_lds_v4i(l), 0, 0);
#else
  unsigned lo = (unsigned)(size_t)l;
  unsigned long long ga = (unsigned long long)(size_t)g;
  asm volatile("global_store_async_from_lds_b128 %1, %0, off"
               :: "v"(lo), "v"(ga) : "memory");
#endif
}

__device__ __forceinline__ void wait_async0() {
#if __has_builtin(__builtin_amdgcn_s_wait_asynccnt)
  __builtin_amdgcn_s_wait_asynccnt(0);
#else
  asm volatile("s_wait_asynccnt 0x0" ::: "memory");
#endif
}

__device__ __forceinline__ void wait_async2() {
#if __has_builtin(__builtin_amdgcn_s_wait_asynccnt)
  __builtin_amdgcn_s_wait_asynccnt(2);
#else
  asm volatile("s_wait_asynccnt 0x2" ::: "memory");
#endif
}

// ---------------------------------------------------------------------------
// WMMA fragment loaders (ISA §7.12.2 layouts), two b128 loads per lane each.
// ---------------------------------------------------------------------------
__device__ __forceinline__ v16bf load_frag_a(const bf16* base, int ld) {
  const int lane = threadIdx.x & 31;
  const int m    = lane & 15;
  const int koff = (lane >> 4) << 3;        // 0 or 8
  const bf16* p = base + m * ld + koff;
  FragU u;
  u.q[0] = *(const uint4*)(p);
  u.q[1] = *(const uint4*)(p + 16);
  return u.v;
}

__device__ __forceinline__ v16bf load_frag_b(const bf16* base, int ld) {
  const int lane = threadIdx.x & 31;
  const int n    = lane & 15;
  const int koff = (lane >> 4) << 3;        // 0 or 8
  const bf16* p = base + n * ld + koff;
  FragU u;
  u.q[0] = *(const uint4*)(p);
  u.q[1] = *(const uint4*)(p + 16);
  return u.v;
}

__device__ __forceinline__ v8f wmma_bf16(v16bf a, v16bf b, v8f c) {
  return __builtin_amdgcn_wmma_f32_16x16x32_bf16(false, a, false, b, (short)0, c,
                                                 false, false);
}

// ---------------------------------------------------------------------------
// Pack bf16 copies of W_ih1 / fc_w (read from L2 during recurrence) + biases.
// ---------------------------------------------------------------------------
__global__ void rnn_pack(const float* __restrict__ wih1, const float* __restrict__ fcw,
                         const float* __restrict__ bih0, const float* __restrict__ bhh0,
                         const float* __restrict__ bih1, const float* __restrict__ bhh1,
                         bf16* __restrict__ p_wih1, bf16* __restrict__ p_fcw,
                         float* __restrict__ b0sum, float* __restrict__ b1sum) {
  int i  = blockIdx.x * blockDim.x + threadIdx.x;
  int st = gridDim.x * blockDim.x;
  for (int j = i; j < H_ * H_;   j += st) p_wih1[j] = (bf16)wih1[j];
  for (int j = i; j < DOUT * H_; j += st) p_fcw[j]  = (bf16)fcw[j];
  for (int j = i; j < H_;        j += st) {
    b0sum[j] = bih0[j] + bhh0[j];
    b1sum[j] = bih1[j] + bhh1[j];
  }
}

// ---------------------------------------------------------------------------
// Layer-0 recurrence: h_t = tanh(x_t W_ih0^T + b0 + h_{t-1} W_hh0^T)
// Single 1024-thread WG (32 waves). Weights LDS-resident (bf16, padded).
// x_{t+1} register-prefetched during step t; h1[t] streamed out with
// global_store_async_from_lds_b128 (2 async ops/thread/step).
// ---------------------------------------------------------------------------
__global__ __launch_bounds__(1024, 1) void rnn_rec0(
    const float* __restrict__ x,      // [B][T][DIN] f32
    const float* __restrict__ wih0,   // [H][DIN]    f32
    const float* __restrict__ whh0,   // [H][H]      f32
    const float* __restrict__ b0sum,  // [H]         f32
    bf16* __restrict__ h1buf)         // [T][B][H]   bf16 out
{
  extern __shared__ char smem[];
  bf16* sWih = (bf16*)smem;              // [H][LDX]
  bf16* sWhh = sWih + H_ * LDX;          // [H][LDH]
  bf16* sX   = sWhh + H_ * LDH;          // [B][LDX]
  bf16* sHa  = sX   + B_ * LDX;          // [B][LDH]
  bf16* sHb  = sHa  + B_ * LDH;          // [B][LDH]

  const int tid = threadIdx.x;
  for (int i = tid; i < H_ * DIN; i += 1024) {
    int r = i >> 7, c = i & 127;
    sWih[r * LDX + c] = (bf16)wih0[i];
  }
  for (int i = tid; i < H_ * H_; i += 1024) {
    int r = i >> 8, c = i & 255;
    sWhh[r * LDH + c] = (bf16)whh0[i];
  }
  for (int i = tid; i < B_ * LDH; i += 1024) sHa[i] = (bf16)0.0f;  // h_{-1}=0

  const int lane = tid & 31;
  const int w    = tid >> 5;             // wave id 0..31
  const int mi   = w >> 3;               // M tile 0..3
  const int n0   = (w & 7) * 2;          // N tile pair {n0, n0+1} of 16
  const int nc0  = n0 * 16 + (lane & 15);
  const int nc1  = nc0 + 16;
  const int mrow = mi * 16 + ((lane >> 4) << 3);
  const float bv0 = b0sum[nc0];
  const float bv1 = b0sum[nc1];

  // x staging map: thread owns 8 consecutive d of row bx (float4 x2 loads).
  const int bx = tid >> 4;               // 0..63
  const int dx = (tid & 15) * 8;         // 0..120
  const float* xrow = x + (size_t)bx * T_ * DIN + dx;
  float4 xr0 = *(const float4*)(xrow + 0);
  float4 xr1 = *(const float4*)(xrow + 4);

  for (int t = 0; t < T_; ++t) {
    // Ensure h-tile async stores from step t-2 (same LDS buffer) are done
    // before anyone overwrites it; combined with the barrier below this
    // orders across all waves (in-order async store completion).
    wait_async2();

    // Commit prefetched x_t to LDS (convert f32->bf16).
    bf16* xd = sX + bx * LDX + dx;
    xd[0] = (bf16)xr0.x; xd[1] = (bf16)xr0.y; xd[2] = (bf16)xr0.z; xd[3] = (bf16)xr0.w;
    xd[4] = (bf16)xr1.x; xd[5] = (bf16)xr1.y; xd[6] = (bf16)xr1.z; xd[7] = (bf16)xr1.w;
    __syncthreads();

    // Prefetch x_{t+1} into registers; latency hidden behind the WMMA chain.
    if (t + 1 < T_) {
      const float* nsrc = xrow + (size_t)(t + 1) * DIN;
      xr0 = *(const float4*)(nsrc + 0);
      xr1 = *(const float4*)(nsrc + 4);
    }

    const bf16* hp = (t & 1) ? sHb : sHa;
    bf16*       hn = (t & 1) ? sHa : sHb;

    v8f c0, c1;
#pragma unroll
    for (int r = 0; r < 8; ++r) { c0[r] = bv0; c1[r] = bv1; }

#pragma unroll
    for (int k = 0; k < DIN; k += 32) {              // input projection, K=128
      v16bf a = load_frag_a(sX + mi * 16 * LDX + k, LDX);
      c0 = wmma_bf16(a, load_frag_b(sWih + (n0    ) * 16 * LDX + k, LDX), c0);
      c1 = wmma_bf16(a, load_frag_b(sWih + (n0 + 1) * 16 * LDX + k, LDX), c1);
    }
#pragma unroll
    for (int k = 0; k < H_; k += 32) {               // hidden projection, K=256
      v16bf a = load_frag_a(hp + mi * 16 * LDH + k, LDH);
      c0 = wmma_bf16(a, load_frag_b(sWhh + (n0    ) * 16 * LDH + k, LDH), c0);
      c1 = wmma_bf16(a, load_frag_b(sWhh + (n0 + 1) * 16 * LDH + k, LDH), c1);
    }

#pragma unroll
    for (int r = 0; r < 8; ++r) {
      int m = mrow + r;
      hn[m * LDH + nc0] = (bf16)tanhf(c0[r]);
      hn[m * LDH + nc1] = (bf16)tanhf(c1[r]);
    }
    __syncthreads();

    // Stream h_t (LDS tile, pad skipped) -> h1buf[t] asynchronously.
    // 2048 16B chunks, 2 per thread.  Chunk c: row c>>5, cols (c&31)*8..+7.
    {
      bf16* outg = h1buf + (size_t)t * B_ * H_;
      int c0i = tid * 2;
      int c1i = c0i + 1;
      async_l2g_b128(hn + (c0i >> 5) * LDH + (c0i & 31) * 8, outg + c0i * 8);
      async_l2g_b128(hn + (c1i >> 5) * LDH + (c1i & 31) * 8, outg + c1i * 8);
    }
  }
  // S_ENDPGM drains ASYNCcnt; kernel boundary orders stores vs rnn_rec1.
}

// ---------------------------------------------------------------------------
// Layer-1 recurrence + final FC.  W_hh1 in LDS; W_ih1 / fc_w B-fragments read
// directly from L2-resident packed bf16.  h1[t] tiles double-buffered in LDS
// via async global->LDS copies issued one step ahead.
// ---------------------------------------------------------------------------
__global__ __launch_bounds__(1024, 1) void rnn_rec1(
    const bf16* __restrict__ h1buf,   // [T][B][H] bf16
    const bf16* __restrict__ wih1,    // [H][H]    bf16 packed
    const float* __restrict__ whh1,   // [H][H]    f32
    const float* __restrict__ b1sum,  // [H]
    const bf16* __restrict__ fcw,     // [DOUT][H] bf16 packed
    const float* __restrict__ fcb,    // [DOUT]
    float* __restrict__ out)          // [B][DOUT] f32
{
  extern __shared__ char smem[];
  bf16* sWhh = (bf16*)smem;            // [H][LDH]
  bf16* sIa  = sWhh + H_ * LDH;        // [B][LDH] input tile buf 0
  bf16* sIb  = sIa  + B_ * LDH;        // [B][LDH] input tile buf 1
  bf16* sHa  = sIb  + B_ * LDH;        // [B][LDH]
  bf16* sHb  = sHa  + B_ * LDH;        // [B][LDH]

  const int tid = threadIdx.x;
  for (int i = tid; i < H_ * H_; i += 1024) {
    int r = i >> 8, c = i & 255;
    sWhh[r * LDH + c] = (bf16)whh1[i];
  }
  for (int i = tid; i < B_ * LDH; i += 1024) sHa[i] = (bf16)0.0f;

  const int lane = tid & 31;
  const int w    = tid >> 5;
  const int mi   = w >> 3;
  const int n0   = (w & 7) * 2;
  const int nc0  = n0 * 16 + (lane & 15);
  const int nc1  = nc0 + 16;
  const int mrow = mi * 16 + ((lane >> 4) << 3);
  const float bv0 = b1sum[nc0];
  const float bv1 = b1sum[nc1];

  // Async staging map: 2048 16B chunks per tile, 2 per thread.
  const int c0i = tid * 2;
  const int c1i = c0i + 1;
  const int l0 = (c0i >> 5) * LDH + (c0i & 31) * 8;   // LDS elem offset chunk 0
  const int l1 = (c1i >> 5) * LDH + (c1i & 31) * 8;

  // Kick off async copy of h1[0] into buffer 0.
  async_g2l_b128(h1buf + c0i * 8, sIa + l0);
  async_g2l_b128(h1buf + c1i * 8, sIa + l1);

  for (int t = 0; t < T_; ++t) {
    bf16* sIn = (t & 1) ? sIb : sIa;

    wait_async0();        // this wave's copies for step t are complete
    __syncthreads();      // all waves' copies visible; h buffers synced

    // Issue async copy of h1[t+1] into the other input buffer.
    if (t + 1 < T_) {
      const bf16* nsrc = h1buf + (size_t)(t + 1) * B_ * H_;
      bf16* nd = (t & 1) ? sIa : sIb;
      async_g2l_b128(nsrc + c0i * 8, nd + l0);
      async_g2l_b128(nsrc + c1i * 8, nd + l1);
    }

    const bf16* hp = (t & 1) ? sHb : sHa;
    bf16*       hn = (t & 1) ? sHa : sHb;

    v8f c0, c1;
#pragma unroll
    for (int r = 0; r < 8; ++r) { c0[r] = bv0; c1[r] = bv1; }

#pragma unroll
    for (int k = 0; k < H_; k += 32) {               // input projection (global B)
      v16bf a = load_frag_a(sIn + mi * 16 * LDH + k, LDH);
      c0 = wmma_bf16(a, load_frag_b(wih1 + (n0    ) * 16 * H_ + k, H_), c0);
      c1 = wmma_bf16(a, load_frag_b(wih1 + (n0 + 1) * 16 * H_ + k, H_), c1);
    }
#pragma unroll
    for (int k = 0; k < H_; k += 32) {               // hidden projection (LDS B)
      v16bf a = load_frag_a(hp + mi * 16 * LDH + k, LDH);
      c0 = wmma_bf16(a, load_frag_b(sWhh + (n0    ) * 16 * LDH + k, LDH), c0);
      c1 = wmma_bf16(a, load_frag_b(sWhh + (n0 + 1) * 16 * LDH + k, LDH), c1);
    }

#pragma unroll
    for (int r = 0; r < 8; ++r) {
      int m = mrow + r;
      hn[m * LDH + nc0] = (bf16)tanhf(c0[r]);
      hn[m * LDH + nc1] = (bf16)tanhf(c1[r]);
    }
    __syncthreads();
  }

  // Final FC: out = h_last @ fc_w^T + fc_b.  T=2048 even -> last hn == sHa.
  {
    const int ni = w & 7;                  // DOUT=128 -> 8 N tiles; 32 tiles
    const int nc = ni * 16 + (lane & 15);
    v8f c;
    const float bv = fcb[nc];
#pragma unroll
    for (int r = 0; r < 8; ++r) c[r] = bv;
#pragma unroll
    for (int k = 0; k < H_; k += 32) {
      v16bf a = load_frag_a(sHa + mi * 16 * LDH + k, LDH);
      c = wmma_bf16(a, load_frag_b(fcw + ni * 16 * H_ + k, H_), c);
    }
#pragma unroll
    for (int r = 0; r < 8; ++r) {
      int m = mrow + r;
      out[m * DOUT + nc] = c[r];
    }
  }
}

// ---------------------------------------------------------------------------
extern "C" void kernel_launch(void* const* d_in, const int* in_sizes, int n_in,
                              void* d_out, int out_size, void* d_ws, size_t ws_size,
                              hipStream_t stream) {
  (void)in_sizes; (void)n_in; (void)out_size; (void)ws_size;

  const float* x    = (const float*)d_in[0];
  const float* wih0 = (const float*)d_in[1];
  const float* whh0 = (const float*)d_in[2];
  const float* bih0 = (const float*)d_in[3];
  const float* bhh0 = (const float*)d_in[4];
  const float* wih1 = (const float*)d_in[5];
  const float* whh1 = (const float*)d_in[6];
  const float* bih1 = (const float*)d_in[7];
  const float* bhh1 = (const float*)d_in[8];
  const float* fcw  = (const float*)d_in[9];
  const float* fcb  = (const float*)d_in[10];
  float* out = (float*)d_out;

  // Workspace layout (bytes):
  //   [0, 131072)           bf16 W_ih1 [256][256]
  //   [131072, 196608)      bf16 fc_w  [128][256]
  //   [196608, 197632)      f32  b0sum [256]
  //   [197632, 198656)      f32  b1sum [256]
  //   [198656, +67,108,864) bf16 h1    [T][B][H]
  char* ws = (char*)d_ws;
  bf16*  p_wih1 = (bf16*)(ws);
  bf16*  p_fcw  = (bf16*)(ws + 131072);
  float* b0sum  = (float*)(ws + 196608);
  float* b1sum  = (float*)(ws + 197632);
  bf16*  h1buf  = (bf16*)(ws + 198656);

  rnn_pack<<<64, 256, 0, stream>>>(wih1, fcw, bih0, bhh0, bih1, bhh1,
                                   p_wih1, p_fcw, b0sum, b1sum);

  const int SMEM0 = (H_ * LDX + H_ * LDH + B_ * LDX + 2 * B_ * LDH) * (int)sizeof(bf16); // 289,792 B
  const int SMEM1 = (H_ * LDH + 4 * B_ * LDH) * (int)sizeof(bf16);                       // 270,336 B
  (void)hipFuncSetAttribute((const void*)rnn_rec0,
                            hipFuncAttributeMaxDynamicSharedMemorySize, SMEM0);
  (void)hipFuncSetAttribute((const void*)rnn_rec1,
                            hipFuncAttributeMaxDynamicSharedMemorySize, SMEM1);

  rnn_rec0<<<1, 1024, SMEM0, stream>>>(x, wih0, whh0, b0sum, h1buf);
  rnn_rec1<<<1, 1024, SMEM1, stream>>>(h1buf, p_wih1, whh1, b1sum, p_fcw, fcb, out);
}
